// Experts_22995254902974
// MI455X (gfx1250) — matrix operand order repaired and checked by reading, not compiled, not run on hardware
//
#include <hip/hip_runtime.h>
#include <hip/hip_bf16.h>
#include <cstdint>

// MI455X / gfx1250: wave32, WMMA 16x16x32 bf16 path + async global->LDS copy.
// Grouped expert FFN: act = silu(hs @ Wg) * (hs @ Wu); out = act @ Wd
// E=8, C=2048, H=2048, I=2816

typedef __attribute__((ext_vector_type(16))) __bf16 v16bf;
typedef __attribute__((ext_vector_type(8)))  float  v8f;

#define E_ 8
#define C_ 2048
#define H_ 2048
#define I_ 2816
#define LDSK 40   // padded row stride (bf16 elems): 80B -> conflict-free b128 frag loads

union FragBF { v16bf v; uint4 q[2]; };
union Pack4  { __bf16 h[4]; uint2 u; };

__device__ __forceinline__ v8f wmma_bf16(const FragBF& a, const FragBF& b, v8f c) {
  return __builtin_amdgcn_wmma_f32_16x16x32_bf16(false, a.v, false, b.v, (short)0, c, false, false);
}

// ---------------------------------------------------------------------------
// Kernel 1: fused gate/up GEMM + SiLU glu. Block tile: 128(M=C) x 128(N of I),
// computing BOTH gate (cols n) and up (cols n+I) tiles. K = H, BK = 32.
// 8 waves: 2(M) x 4(N); wave tile 64x32 per matrix -> 4x2 accum x2 matrices
// = 16 WMMA per k-step from 8 fragment loads.
// ---------------------------------------------------------------------------
__global__ __launch_bounds__(256) void gateup_silu_kernel(
    const float* __restrict__ hs,   // [E, C, H]
    const float* __restrict__ gw,   // [E, H, 2I]
    __bf16* __restrict__ act)       // [E, C, I]  (bf16 workspace)
{
  __shared__ __align__(16) __bf16 As[2][128][LDSK];
  __shared__ __align__(16) __bf16 Bg[2][128][LDSK];   // transposed [n][k]
  __shared__ __align__(16) __bf16 Bu[2][128][LDSK];   // transposed [n][k]

  const int tid   = threadIdx.x;
  const int e     = blockIdx.z;
  const int cBase = blockIdx.y * 128;
  const int nBase = blockIdx.x * 128;

  const float* hsE = hs + ((size_t)e * C_ + cBase) * H_;
  const float* gwE = gw + (size_t)e * H_ * (2 * I_);

  const int lane = tid & 31;
  const int wave = tid >> 5;
  const int wm   = wave >> 2;   // 0..1 -> M offset wm*64
  const int wn   = wave & 3;    // 0..3 -> N offset wn*32
  const int r    = lane & 15;
  const int hi   = lane >> 4;

  v8f accG[4][2], accU[4][2];
  {
    v8f z = {};
    #pragma unroll
    for (int i = 0; i < 4; ++i)
      #pragma unroll
      for (int j = 0; j < 2; ++j) { accG[i][j] = z; accU[i][j] = z; }
  }

  const int NK = H_ / 32;

  auto load_tiles = [&](int kt, int buf) {
    const int k0 = kt * 32;
    // A: 128 rows x 32 K of f32 -> bf16. 1024 float4 chunks / 256 thr = 4 ea.
    #pragma unroll
    for (int i = 0; i < 4; ++i) {
      const int idx = tid + i * 256;
      const int row = idx >> 3;
      const int q   = (idx & 7) * 4;
      const float4 f = *reinterpret_cast<const float4*>(hsE + (size_t)row * H_ + k0 + q);
      Pack4 p;
      p.h[0] = (__bf16)f.x; p.h[1] = (__bf16)f.y;
      p.h[2] = (__bf16)f.z; p.h[3] = (__bf16)f.w;
      *reinterpret_cast<uint2*>(&As[buf][row][q]) = p.u;
    }
    // B gate/up: 32 K-rows x 128 N of f32 each, store transposed [n][k].
    // 1024 float4 chunks per matrix / 256 thr = 4 ea.
    #pragma unroll
    for (int i = 0; i < 4; ++i) {
      const int idx = tid + i * 256;
      const int k   = idx >> 5;
      const int n   = (idx & 31) * 4;
      const float* rowp = gwE + (size_t)(k0 + k) * (2 * I_) + nBase + n;
      const float4 fg = *reinterpret_cast<const float4*>(rowp);
      const float4 fu = *reinterpret_cast<const float4*>(rowp + I_);
      Bg[buf][n + 0][k] = (__bf16)fg.x; Bg[buf][n + 1][k] = (__bf16)fg.y;
      Bg[buf][n + 2][k] = (__bf16)fg.z; Bg[buf][n + 3][k] = (__bf16)fg.w;
      Bu[buf][n + 0][k] = (__bf16)fu.x; Bu[buf][n + 1][k] = (__bf16)fu.y;
      Bu[buf][n + 2][k] = (__bf16)fu.z; Bu[buf][n + 3][k] = (__bf16)fu.w;
    }
  };

  load_tiles(0, 0);
  __syncthreads();

  for (int kt = 0; kt < NK; ++kt) {
    const int buf = kt & 1;
    if (kt + 1 < NK) load_tiles(kt + 1, buf ^ 1);

    // Fragment loads (ISA 16-bit layouts):
    // A: lane m=r; elems 0..7 = K[hi*8..], elems 8..15 = K[16+hi*8..]
    // B(transposed in LDS): lane n=r; elems 0..7 = K[hi*8..], elems 8..15 = K[16+hi*8..]
    FragBF a[4], bg[2], bu[2];
    #pragma unroll
    for (int im = 0; im < 4; ++im) {
      const __bf16* ap = &As[buf][wm * 64 + im * 16 + r][hi * 8];
      a[im].q[0] = *reinterpret_cast<const uint4*>(ap);
      a[im].q[1] = *reinterpret_cast<const uint4*>(ap + 16);
    }
    #pragma unroll
    for (int in = 0; in < 2; ++in) {
      const __bf16* gp = &Bg[buf][wn * 32 + in * 16 + r][hi * 8];
      bg[in].q[0] = *reinterpret_cast<const uint4*>(gp);
      bg[in].q[1] = *reinterpret_cast<const uint4*>(gp + 16);
      const __bf16* up = &Bu[buf][wn * 32 + in * 16 + r][hi * 8];
      bu[in].q[0] = *reinterpret_cast<const uint4*>(up);
      bu[in].q[1] = *reinterpret_cast<const uint4*>(up + 16);
    }
    #pragma unroll
    for (int im = 0; im < 4; ++im)
      #pragma unroll
      for (int in = 0; in < 2; ++in) {
        accG[im][in] = wmma_bf16(a[im], bg[in], accG[im][in]);
        accU[im][in] = wmma_bf16(a[im], bu[in], accU[im][in]);
      }
    __syncthreads();
  }

  // Epilogue: silu(gate) * up, write bf16 activation.
  // C/D layout: VGPR j -> M = j + hi*8, N = r.
  __bf16* actE = act + (size_t)e * C_ * I_;
  #pragma unroll
  for (int im = 0; im < 4; ++im)
    #pragma unroll
    for (int in = 0; in < 2; ++in) {
      const int n  = nBase + wn * 32 + in * 16 + r;
      const int m0 = cBase + wm * 64 + im * 16 + hi * 8;
      #pragma unroll
      for (int j = 0; j < 8; ++j) {
        const float g = accG[im][in][j];
        const float u = accU[im][in][j];
        const float s = g / (1.0f + __expf(-g));
        actE[(size_t)(m0 + j) * I_ + n] = (__bf16)(s * u);
      }
    }
}

// ---------------------------------------------------------------------------
// Kernel 2: down projection. out[E,C,H] = act[E,C,I](bf16) @ down_w[E,I,H].
// Block tile 256x128, BK=32. 8 waves: 4(M) x 2(N); wave tile 64x64 -> 4x4
// = 16 WMMA per k-step from 8 fragment loads.
// A tile is pure bf16 copy -> GLOBAL_LOAD_ASYNC_TO_LDS_B128 (ASYNCcnt path).
// ---------------------------------------------------------------------------
__global__ __launch_bounds__(256) void down_kernel(
    const __bf16* __restrict__ act,  // [E, C, I]
    const float* __restrict__ dw,    // [E, I, H]
    float* __restrict__ out)         // [E, C, H]
{
  __shared__ __align__(16) __bf16 As[2][256][LDSK];
  __shared__ __align__(16) __bf16 Bs[2][128][LDSK];  // transposed [n][k]

  const int tid   = threadIdx.x;
  const int e     = blockIdx.z;
  const int cBase = blockIdx.y * 256;
  const int nBase = blockIdx.x * 128;

  const __bf16* actE = act + ((size_t)e * C_ + cBase) * I_;
  const float*  dwE  = dw + (size_t)e * I_ * H_;
  const uint64_t aBase = (uint64_t)(uintptr_t)actE;

  const int lane = tid & 31;
  const int wave = tid >> 5;
  const int wm   = wave >> 1;   // 0..3 -> M offset wm*64
  const int wn   = wave & 1;    // 0..1 -> N offset wn*64
  const int r    = lane & 15;
  const int hi   = lane >> 4;

  v8f acc[4][4];
  {
    v8f z = {};
    #pragma unroll
    for (int i = 0; i < 4; ++i)
      #pragma unroll
      for (int j = 0; j < 4; ++j) acc[i][j] = z;
  }

  const int NK = I_ / 32;   // 88

  auto load_tiles = [&](int kt, int buf) {
    const int k0 = kt * 32;
    // A: 256 rows x 32 K bf16 -> async copy straight into LDS (no VGPR pass).
    // 1024 chunks of 8 bf16 (16B) / 256 thr = 4 ea.
    #pragma unroll
    for (int i = 0; i < 4; ++i) {
      const int idx = tid + i * 256;
      const int row = idx >> 2;
      const int cq  = (idx & 3) * 8;
      const uint32_t goff = (uint32_t)((row * I_ + k0 + cq) * 2);
      const uint32_t ldsa = (uint32_t)(uintptr_t)(&As[buf][row][cq]);
      asm volatile("global_load_async_to_lds_b128 %0, %1, %2"
                   :: "v"(ldsa), "v"(goff), "s"(aBase) : "memory");
    }
    // B: 32 K-rows x 128 N of f32, convert + store transposed [n][k].
    // 1024 float4 chunks / 256 thr = 4 ea.
    #pragma unroll
    for (int i = 0; i < 4; ++i) {
      const int idx = tid + i * 256;
      const int k   = idx >> 5;
      const int n   = (idx & 31) * 4;
      const float4 f = *reinterpret_cast<const float4*>(dwE + (size_t)(k0 + k) * H_ + nBase + n);
      Bs[buf][n + 0][k] = (__bf16)f.x; Bs[buf][n + 1][k] = (__bf16)f.y;
      Bs[buf][n + 2][k] = (__bf16)f.z; Bs[buf][n + 3][k] = (__bf16)f.w;
    }
  };

  load_tiles(0, 0);
  asm volatile("s_wait_asynccnt 0x0" ::: "memory");
  __syncthreads();

  for (int kt = 0; kt < NK; ++kt) {
    const int buf = kt & 1;
    if (kt + 1 < NK) load_tiles(kt + 1, buf ^ 1);

    FragBF a[4], b[4];
    #pragma unroll
    for (int im = 0; im < 4; ++im) {
      const __bf16* ap = &As[buf][wm * 64 + im * 16 + r][hi * 8];
      a[im].q[0] = *reinterpret_cast<const uint4*>(ap);
      a[im].q[1] = *reinterpret_cast<const uint4*>(ap + 16);
    }
    #pragma unroll
    for (int in = 0; in < 4; ++in) {
      const __bf16* bp = &Bs[buf][wn * 64 + in * 16 + r][hi * 8];
      b[in].q[0] = *reinterpret_cast<const uint4*>(bp);
      b[in].q[1] = *reinterpret_cast<const uint4*>(bp + 16);
    }
    #pragma unroll
    for (int im = 0; im < 4; ++im)
      #pragma unroll
      for (int in = 0; in < 4; ++in)
        acc[im][in] = wmma_bf16(a[im], b[in], acc[im][in]);

    // Async A-tile writes for the next buffer must land before anyone reads it.
    asm volatile("s_wait_asynccnt 0x0" ::: "memory");
    __syncthreads();
  }

  float* outE = out + (size_t)e * C_ * H_;
  #pragma unroll
  for (int im = 0; im < 4; ++im)
    #pragma unroll
    for (int in = 0; in < 4; ++in) {
      const int n  = nBase + wn * 64 + in * 16 + r;
      const int m0 = cBase + wm * 64 + im * 16 + hi * 8;
      #pragma unroll
      for (int j = 0; j < 8; ++j)
        outE[(size_t)(m0 + j) * H_ + n] = acc[im][in][j];
    }
}

// ---------------------------------------------------------------------------
extern "C" void kernel_launch(void* const* d_in, const int* in_sizes, int n_in,
                              void* d_out, int out_size, void* d_ws, size_t ws_size,
                              hipStream_t stream) {
  (void)in_sizes; (void)n_in; (void)out_size; (void)ws_size;
  const float* hs = (const float*)d_in[0];   // [E, C, H] f32
  const float* gw = (const float*)d_in[1];   // [E, H, 2I] f32
  const float* dw = (const float*)d_in[2];   // [E, I, H] f32
  float* out = (float*)d_out;                // [E, C, H] f32
  __bf16* act = (__bf16*)d_ws;               // [E, C, I] bf16 (92.3 MB scratch)

  // GEMM1 + SiLU-GLU fused
  gateup_silu_kernel<<<dim3(I_ / 128, C_ / 128, E_), 256, 0, stream>>>(hs, gw, act);
  // GEMM2 (async A staging)
  down_kernel<<<dim3(H_ / 128, C_ / 256, E_), 256, 0, stream>>>(act, dw, out);
}
